// SelectiveSSM_10900626997667
// MI455X (gfx1250) — matrix operand order repaired and checked by reading, not compiled, not run on hardware
//
#include <hip/hip_runtime.h>
#include <hip/hip_bf16.h>

// ---- problem constants (match reference) ----
#define BATCH   2
#define SEQ     2048
#define DMODEL  1024
#define DINNER  2048          // EXPAND * DMODEL
#define NSTATE  16
#define KCONV   4
#define NTOK    (BATCH * SEQ)       // 4096 tokens

typedef _Float16 half8  __attribute__((ext_vector_type(8)));
typedef _Float16 half16 __attribute__((ext_vector_type(16)));
typedef float    f32x8  __attribute__((ext_vector_type(8)));

// --------------------------------------------------------------------------
// fp32 -> fp16 conversion (prep for WMMA operands)
// --------------------------------------------------------------------------
__global__ void cvt_f32_to_f16(const float* __restrict__ in,
                               _Float16* __restrict__ out, int n) {
    int i = blockIdx.x * blockDim.x + threadIdx.x;
    if (i < n) out[i] = (_Float16)in[i];
}

// --------------------------------------------------------------------------
// WMMA GEMM: C[M,N] = A[M,K] * B[N,K]^T   (A,B f16 row-major, C f32 row-major)
// one wave -> 16x64 output strip (4 x v_wmma_f32_16x16x32_f16 per K-step,
// A fragment reused across the 4 N-subtiles). K % 32 == 0, N % 64 == 0.
// --------------------------------------------------------------------------
__global__ void wmma_gemm_f16f32(const _Float16* __restrict__ A,
                                 const _Float16* __restrict__ Bw,
                                 float* __restrict__ C,
                                 int M, int N, int K) {
    const int lane  = threadIdx.x & 31;
    const int wave  = threadIdx.x >> 5;
    const int mTile = blockIdx.y * 4 + wave;       // 16 rows per tile
    const int m0    = mTile * 16;
    const int nBase = blockIdx.x * 64;
    if (m0 >= M) return;                           // wave-uniform

    const int hi   = lane >> 4;                    // half-wave select
    const int mRow = m0 + (lane & 15);             // A: M striped in lanes 0-15
    const int aOff = hi * 16;                      // A: K sub-block per half-wave
    const int nCol = lane & 15;                    // B: N striped in lanes 0-15
    const int bK   = hi * 16;                      // B: K sub-block per half-wave

    const _Float16* Arow = A + (size_t)mRow * K;

    f32x8 acc0 = (f32x8)0.f, acc1 = (f32x8)0.f, acc2 = (f32x8)0.f, acc3 = (f32x8)0.f;

    for (int k0 = 0; k0 < K; k0 += 32) {
        // A 16x32 fragment: 16 contiguous halfs per lane (two 128-bit loads)
        half8 alo = *(const half8*)(Arow + k0 + aOff);
        half8 ahi = *(const half8*)(Arow + k0 + aOff + 8);
        half16 a;
        #pragma unroll
        for (int i = 0; i < 8; ++i) { a[i] = alo[i]; a[8 + i] = ahi[i]; }

        // B 32x16 fragments: 16 contiguous halfs from weight row (N,K) row-major
        const _Float16* B0 = Bw + (size_t)(nBase +  0 + nCol) * K + k0 + bK;
        const _Float16* B1 = Bw + (size_t)(nBase + 16 + nCol) * K + k0 + bK;
        const _Float16* B2 = Bw + (size_t)(nBase + 32 + nCol) * K + k0 + bK;
        const _Float16* B3 = Bw + (size_t)(nBase + 48 + nCol) * K + k0 + bK;
        half16 b0 = *(const half16*)B0;
        half16 b1 = *(const half16*)B1;
        half16 b2 = *(const half16*)B2;
        half16 b3 = *(const half16*)B3;

        acc0 = __builtin_amdgcn_wmma_f32_16x16x32_f16(false, a, false, b0, (short)0, acc0, false, false);
        acc1 = __builtin_amdgcn_wmma_f32_16x16x32_f16(false, a, false, b1, (short)0, acc1, false, false);
        acc2 = __builtin_amdgcn_wmma_f32_16x16x32_f16(false, a, false, b2, (short)0, acc2, false, false);
        acc3 = __builtin_amdgcn_wmma_f32_16x16x32_f16(false, a, false, b3, (short)0, acc3, false, false);
    }

    // D layout: VGPR r -> row m0 + r + hi*8, col = nSub + (lane&15)
    const int rBase = m0 + hi * 8;
    #pragma unroll
    for (int r = 0; r < 8; ++r) {
        size_t row = (size_t)(rBase + r) * N;
        C[row + nBase +  0 + nCol] = acc0[r];
        C[row + nBase + 16 + nCol] = acc1[r];
        C[row + nBase + 32 + nCol] = acc2[r];
        C[row + nBase + 48 + nCol] = acc3[r];
    }
}

// --------------------------------------------------------------------------
// causal depthwise conv1d (k=4, left-pad 3) + bias + SiLU  : xz[:, :DINNER] -> u
// --------------------------------------------------------------------------
__global__ void conv_silu_kernel(const float* __restrict__ xz,
                                 const float* __restrict__ cw,
                                 const float* __restrict__ cb,
                                 float* __restrict__ u) {
    int idx = blockIdx.x * blockDim.x + threadIdx.x;   // (tok, d), d fastest
    if (idx >= NTOK * DINNER) return;
    int d   = idx & (DINNER - 1);
    int tok = idx >> 11;
    int t   = tok & (SEQ - 1);
    float s = cb[d];
    #pragma unroll
    for (int j = 0; j < KCONV; ++j) {
        int tt = t - (KCONV - 1) + j;
        if (tt >= 0)
            s += cw[d * KCONV + j] * xz[(size_t)(tok - (KCONV - 1) + j) * (2 * DINNER) + d];
    }
    u[idx] = s / (1.f + __expf(-s));   // SiLU
}

// --------------------------------------------------------------------------
// x_dbl = u @ W_x^T : one wave per (token, e) of the 33 outputs; split into
// dlt scalar (e==0), Bc[token,16], Cc[token,16]
// --------------------------------------------------------------------------
__global__ void xdbl_kernel(const float* __restrict__ u,
                            const float* __restrict__ Wx,
                            float* __restrict__ dlt,
                            float* __restrict__ Bc,
                            float* __restrict__ Cc) {
    int gw   = (blockIdx.x * blockDim.x + threadIdx.x) >> 5;
    int lane = threadIdx.x & 31;
    if (gw >= NTOK * (2 * NSTATE + 1)) return;
    int tok = gw / (2 * NSTATE + 1);
    int e   = gw % (2 * NSTATE + 1);
    const float* urow = u  + (size_t)tok * DINNER;
    const float* wrow = Wx + (size_t)e   * DINNER;
    float s = 0.f;
    for (int d = lane; d < DINNER; d += 32) s += urow[d] * wrow[d];
    #pragma unroll
    for (int off = 16; off >= 1; off >>= 1) s += __shfl_xor(s, off, 32);
    if (lane == 0) {
        if (e == 0)                dlt[tok] = s;
        else if (e <= NSTATE)      Bc[tok * NSTATE + (e - 1)] = s;
        else                       Cc[tok * NSTATE + (e - 1 - NSTATE)] = s;
    }
}

// --------------------------------------------------------------------------
// delta[tok,d] = softplus(dlt[tok] * W_dt[d] + b_dt[d])
// --------------------------------------------------------------------------
__global__ void delta_kernel(const float* __restrict__ dlt,
                             const float* __restrict__ Wdt,
                             const float* __restrict__ bdt,
                             float* __restrict__ delta) {
    int idx = blockIdx.x * blockDim.x + threadIdx.x;
    if (idx >= NTOK * DINNER) return;
    int d   = idx & (DINNER - 1);
    int tok = idx >> 11;
    float x = dlt[tok] * Wdt[d] + bdt[d];
    delta[idx] = (x > 20.f) ? x : log1pf(__expf(x));
}

// --------------------------------------------------------------------------
// selective scan: wave32 = 2 channels x 16 states. lane -> (channel, n).
// h_{t} = exp(dt*A)*h_{t-1} + (dt*u)*B_t ;  y = <h, C_t> ; yg = y * silu(z)
// 16-lane shfl_xor tree reduces over the state dim. Output f16 for out_proj.
// --------------------------------------------------------------------------
__global__ void scan_kernel(const float* __restrict__ u,
                            const float* __restrict__ delta,
                            const float* __restrict__ Bc,
                            const float* __restrict__ Cc,
                            const float* __restrict__ Am,
                            const float* __restrict__ xz,
                            _Float16* __restrict__ yg) {
    const int lane = threadIdx.x & 31;
    const int wave = threadIdx.x >> 5;
    const int cp   = blockIdx.x * 8 + wave;      // channel pair, 0..2047
    const int c    = cp * 2 + (lane >> 4);       // channel 0..(B*DINNER-1)
    const int n    = lane & 15;
    const int b    = c >> 11;                    // / DINNER
    const int d    = c & (DINNER - 1);

    const float Av = Am[d * NSTATE + n];
    float h = 0.f;
    const size_t tokBase = (size_t)b * SEQ;

    for (int t = 0; t < SEQ; ++t) {
        const size_t tok = tokBase + t;
        float dt = delta[tok * DINNER + d];
        float uv = u    [tok * DINNER + d];
        float Bn = Bc   [tok * NSTATE + n];
        float Cn = Cc   [tok * NSTATE + n];

        h = __expf(dt * Av) * h + (dt * uv) * Bn;

        float p = h * Cn;
        p += __shfl_xor(p, 1, 32);
        p += __shfl_xor(p, 2, 32);
        p += __shfl_xor(p, 4, 32);
        p += __shfl_xor(p, 8, 32);

        if (n == 0) {
            float z = xz[tok * (2 * DINNER) + DINNER + d];
            float g = z / (1.f + __expf(-z));
            yg[tok * DINNER + d] = (_Float16)(p * g);
        }
    }
}

// --------------------------------------------------------------------------
extern "C" void kernel_launch(void* const* d_in, const int* in_sizes, int n_in,
                              void* d_out, int out_size, void* d_ws, size_t ws_size,
                              hipStream_t stream) {
    const float* x      = (const float*)d_in[0];   // (B,L,DMODEL)
    const float* W_in   = (const float*)d_in[1];   // (2*DINNER, DMODEL)
    const float* conv_w = (const float*)d_in[2];   // (DINNER,1,4)
    const float* conv_b = (const float*)d_in[3];   // (DINNER,)
    const float* A      = (const float*)d_in[4];   // (DINNER,NSTATE)
    const float* W_x    = (const float*)d_in[5];   // (2N+1, DINNER)
    const float* W_dt   = (const float*)d_in[6];   // (DINNER,1)
    const float* b_dt   = (const float*)d_in[7];   // (DINNER,)
    const float* W_out  = (const float*)d_in[8];   // (DMODEL, DINNER)
    float* out = (float*)d_out;                    // (B,L,DMODEL)

    // ---- workspace carve-up (all sizes 256B aligned) ----
    char* ws = (char*)d_ws;
    size_t off = 0;
    auto take = [&](size_t bytes) { char* p = ws + off; off += (bytes + 255) & ~(size_t)255; return p; };

    _Float16* x_h    = (_Float16*)take((size_t)NTOK * DMODEL * 2);          // 8.4 MB
    _Float16* win_h  = (_Float16*)take((size_t)2 * DINNER * DMODEL * 2);    // 8.4 MB
    _Float16* wout_h = (_Float16*)take((size_t)DMODEL * DINNER * 2);        // 4.2 MB
    float*    xz     = (float*)   take((size_t)NTOK * 2 * DINNER * 4);      // 67 MB
    float*    u      = (float*)   take((size_t)NTOK * DINNER * 4);          // 33.6 MB
    float*    dlt    = (float*)   take((size_t)NTOK * 4);
    float*    Bc     = (float*)   take((size_t)NTOK * NSTATE * 4);
    float*    Cc     = (float*)   take((size_t)NTOK * NSTATE * 4);
    float*    delta  = (float*)   take((size_t)NTOK * DINNER * 4);          // 33.6 MB
    _Float16* yg_h   = (_Float16*)take((size_t)NTOK * DINNER * 2);          // 16.8 MB
    (void)ws_size; (void)in_sizes; (void)n_in; (void)out_size;

    // 1) f16 operand prep
    {
        int n1 = NTOK * DMODEL;
        cvt_f32_to_f16<<<(n1 + 255) / 256, 256, 0, stream>>>(x, x_h, n1);
        int n2 = 2 * DINNER * DMODEL;
        cvt_f32_to_f16<<<(n2 + 255) / 256, 256, 0, stream>>>(W_in, win_h, n2);
        int n3 = DMODEL * DINNER;
        cvt_f32_to_f16<<<(n3 + 255) / 256, 256, 0, stream>>>(W_out, wout_h, n3);
    }

    // 2) in_proj GEMM: xz = x @ W_in^T   (M=4096, N=4096, K=1024)
    {
        dim3 grid(2 * DINNER / 64, NTOK / 64);   // (64, 64); 4 waves/block, 4 M-tiles
        wmma_gemm_f16f32<<<grid, 128, 0, stream>>>(x_h, win_h, xz,
                                                   NTOK, 2 * DINNER, DMODEL);
    }

    // 3) causal depthwise conv + SiLU -> u
    conv_silu_kernel<<<(NTOK * DINNER) / 256, 256, 0, stream>>>(xz, conv_w, conv_b, u);

    // 4) x_dbl projection -> dlt, Bc, Cc   (one wave per (token, e))
    {
        int waves  = NTOK * (2 * NSTATE + 1);
        int blocks = (waves * 32 + 255) / 256;
        xdbl_kernel<<<blocks, 256, 0, stream>>>(u, W_x, dlt, Bc, Cc);
    }

    // 5) delta = softplus(dlt * W_dt + b_dt)
    delta_kernel<<<(NTOK * DINNER) / 256, 256, 0, stream>>>(dlt, W_dt, b_dt, delta);

    // 6) selective scan + gate -> yg (f16)
    scan_kernel<<<(BATCH * DINNER) / 16, 256, 0, stream>>>(u, delta, Bc, Cc, A, xz, yg_h);

    // 7) out_proj GEMM: out = yg @ W_out^T   (M=4096, N=1024, K=2048)
    {
        dim3 grid(DMODEL / 64, NTOK / 64);       // (16, 64)
        wmma_gemm_f16f32<<<grid, 128, 0, stream>>>(yg_h, wout_h, out,
                                                   NTOK, DMODEL, DINNER);
    }
}
